// TrafficGNN_31971736551692
// MI455X (gfx1250) — matrix operand order repaired and checked by reading, not compiled, not run on hardware
//
#include <hip/hip_runtime.h>
#include <math.h>

// Problem constants (from reference)
#define NN   50000
#define EE   800000
#define ETOT 850000     // EE + NN self-loops
#define HHID 256        // H * HID = 4 * 64
#define NHEAD 4
#define CHID 64

// ---------------- WMMA types ----------------
typedef __attribute__((ext_vector_type(16))) __bf16       v16bf;
typedef __attribute__((ext_vector_type(8)))  float        v8f;
typedef __attribute__((ext_vector_type(4)))  unsigned int v4u;

union FragBF { v4u u[2]; v16bf v; };

__device__ __forceinline__ unsigned short f2bf(float f) {
  unsigned u = __float_as_uint(f);
  if ((u & 0x7f800000u) == 0x7f800000u) return (unsigned short)(u >> 16); // Inf/NaN
  u += 0x7fffu + ((u >> 16) & 1u);  // round to nearest even
  return (unsigned short)(u >> 16);
}

// ---------------- packing kernels ----------------
__global__ void pack_bf16_kernel(const float* __restrict__ in,
                                 unsigned short* __restrict__ outp, int n) {
  int t = blockIdx.x * blockDim.x + threadIdx.x;
  if (t < n) outp[t] = f2bf(in[t]);
}

// W is [K, 256] row-major fp32 -> wb is [256, K] (i.e. column of W contiguous) bf16
__global__ void pack_wt_kernel(const float* __restrict__ Wm,
                               unsigned short* __restrict__ wb, int K) {
  int t = blockIdx.x * blockDim.x + threadIdx.x;  // over 256*K
  if (t >= 256 * K) return;
  int col = t / K, k = t - col * K;
  wb[t] = f2bf(Wm[k * 256 + col]);
}

// ---------------- WMMA GEMM: C[M,256] = A[M,K](bf16) * B[K,256](bf16, stored B^T) ----------------
// One wave -> one 16x16 tile of C. K in steps of 32 via v_wmma_f32_16x16x32_bf16.
__global__ __launch_bounds__(128) void gemm_wmma_bf16_kernel(
    const unsigned short* __restrict__ A,   // [M x K] bf16 row-major
    const unsigned short* __restrict__ Bt,  // [256 x K] bf16 (W transposed)
    float* __restrict__ C,                  // [M x 256] f32
    int M, int K) {
  const int lane = threadIdx.x & 31;
  const int wid  = blockIdx.x * (blockDim.x >> 5) + (threadIdx.x >> 5);
  const int tilesN = HHID >> 4;            // 16
  const int tm = wid / tilesN;
  const int tn = wid - tm * tilesN;
  if (tm * 16 >= M) return;

  const int r  = lane & 15;
  const int hi = lane >> 4;                // 0 or 1 (lane half)

  // A fragment (16x32, 16-bit): lanes 0-15 hold row r, K = 0..15;
  // lanes 16-31 hold row r, K = 16..31.
  const unsigned short* arow = A  + (size_t)(tm * 16 + r) * K;
  // B fragment (32x16, 16-bit): lanes 0-15 hold col r, K = 0..15;
  // lanes 16-31 hold col r, K = 16..31. Bt row = column of W (contiguous K).
  const unsigned short* bcol = Bt + (size_t)(tn * 16 + r) * K;

  v8f acc = {};
  for (int k0 = 0; k0 < K; k0 += 32) {
    FragBF fa, fb;
    const unsigned short* ap = arow + k0 + hi * 16;
    fa.u[0] = *(const v4u*)(ap);           // K +0..7  of this half
    fa.u[1] = *(const v4u*)(ap + 8);       // K +8..15 of this half
    const unsigned short* bp = bcol + k0 + hi * 16;
    fb.u[0] = *(const v4u*)(bp);           // K +0..7  of this half
    fb.u[1] = *(const v4u*)(bp + 8);       // K +8..15 of this half
    acc = __builtin_amdgcn_wmma_f32_16x16x32_bf16(
        /*neg_a=*/false, fa.v, /*neg_b=*/false, fb.v,
        /*c_mod=*/(short)0, acc, /*reuse_a=*/false, /*reuse_b=*/false);
  }
  // C/D layout: VGPR p -> row tm*16 + p + hi*8, col tn*16 + (lane&15)
  float* crow = C + (size_t)(tm * 16 + hi * 8) * HHID + tn * 16 + r;
#pragma unroll
  for (int p = 0; p < 8; ++p) crow[(size_t)p * HHID] = acc[p];
}

// ---------------- attention coefficients: es/ed = <h[n,h,:], a_{s,d}[h,:]> ----------------
__global__ void attn_coef_kernel(const float* __restrict__ h,
                                 const float* __restrict__ a_s,
                                 const float* __restrict__ a_d,
                                 float* __restrict__ es, float* __restrict__ ed) {
  int t = blockIdx.x * blockDim.x + threadIdx.x;    // over NN*NHEAD
  if (t >= NN * NHEAD) return;
  int node = t >> 2, hd = t & 3;
  const float* hp  = h   + (size_t)node * HHID + hd * CHID;
  const float* asw = a_s + hd * CHID;
  const float* adw = a_d + hd * CHID;
  float s = 0.f, d = 0.f;
#pragma unroll 8
  for (int c = 0; c < CHID; ++c) { float v = hp[c]; s += v * asw[c]; d += v * adw[c]; }
  es[t] = s; ed[t] = d;
}

// ---------------- per-layer state init (accum=0, m=-inf, z=0) ----------------
__global__ void init_kernel(float* __restrict__ accum, float* __restrict__ m,
                            float* __restrict__ z) {
  int t = blockIdx.x * blockDim.x + threadIdx.x;    // over NN*HHID
  if (t < NN * HHID) accum[t] = 0.f;
  if (t < NN * NHEAD) { m[t] = -INFINITY; z[t] = 0.f; }
}

__device__ __forceinline__ void edge_sd(const int* __restrict__ ei, int e, int& s, int& d) {
  if (e < EE) { s = ei[e]; d = ei[EE + e]; } else { s = d = e - EE; }
}

// ---------------- edge logits + segment max ----------------
__global__ void edge_logits_kernel(const int* __restrict__ ei,
                                   const float* __restrict__ es,
                                   const float* __restrict__ ed,
                                   float* __restrict__ elog, float* __restrict__ m) {
  int t = blockIdx.x * blockDim.x + threadIdx.x;    // over ETOT*NHEAD
  if (t >= ETOT * NHEAD) return;
  int e = t >> 2, hd = t & 3, s, d;
  edge_sd(ei, e, s, d);
  float v = es[s * NHEAD + hd] + ed[d * NHEAD + hd];
  v = v > 0.f ? v : 0.2f * v;                       // leaky_relu(0.2)
  elog[t] = v;
  float* addr = &m[d * NHEAD + hd];                 // float atomic max via int/uint trick
  if (v >= 0.f) atomicMax((int*)addr, __float_as_int(v));
  else          atomicMin((unsigned int*)addr, __float_as_uint(v));
}

// ---------------- exp(e - m[dst]) + segment sum ----------------
__global__ void edge_exp_kernel(const int* __restrict__ ei,
                                const float* __restrict__ m,
                                float* __restrict__ elog, float* __restrict__ z) {
  int t = blockIdx.x * blockDim.x + threadIdx.x;
  if (t >= ETOT * NHEAD) return;
  int e = t >> 2, hd = t & 3, s, d;
  edge_sd(ei, e, s, d);
  float w = expf(elog[t] - m[d * NHEAD + hd]);
  elog[t] = w;
  atomicAdd(&z[d * NHEAD + hd], w);
}

// ---------------- alpha = w / z[dst] ----------------
__global__ void edge_norm_kernel(const int* __restrict__ ei,
                                 const float* __restrict__ z,
                                 float* __restrict__ elog) {
  int t = blockIdx.x * blockDim.x + threadIdx.x;
  if (t >= ETOT * NHEAD) return;
  int e = t >> 2, hd = t & 3, s, d;
  edge_sd(ei, e, s, d);
  elog[t] = elog[t] / z[d * NHEAD + hd];
}

// ---------------- scatter: accum[dst,:] += alpha * h[src,:] ----------------
// block = edge, thread = channel (fully coalesced gather + f32 atomic scatter)
__global__ __launch_bounds__(256) void scatter_kernel(const int* __restrict__ ei,
                                                      const float* __restrict__ alpha,
                                                      const float* __restrict__ h,
                                                      float* __restrict__ accum) {
  int e = blockIdx.x;
  int c = threadIdx.x;                              // 0..255
  int s, d;
  edge_sd(ei, e, s, d);
  float a = alpha[(size_t)e * NHEAD + (c >> 6)];
  atomicAdd(&accum[(size_t)d * HHID + c], a * h[(size_t)s * HHID + c]);
}

// ---------------- head mean + bias + exact GELU ----------------
__global__ void finalize_kernel(const float* __restrict__ accum,
                                const float* __restrict__ b,
                                float* __restrict__ xc) {
  int t = blockIdx.x * blockDim.x + threadIdx.x;    // over NN*CHID
  if (t >= NN * CHID) return;
  int node = t >> 6, c = t & 63;
  const float* ap = accum + (size_t)node * HHID + c;
  float v = 0.25f * (ap[0] + ap[64] + ap[128] + ap[192]) + b[c];
  xc[t] = 0.5f * v * (1.f + erff(v * 0.70710678118654752f));
}

// ---------------- final fc + sigmoid ----------------
__global__ void fc_kernel(const float* __restrict__ xc, const float* __restrict__ fcW,
                          const float* __restrict__ fcB, float* __restrict__ out) {
  int n = blockIdx.x * blockDim.x + threadIdx.x;
  if (n >= NN) return;
  const float* r = xc + (size_t)n * CHID;
  float v = fcB[0];
#pragma unroll 8
  for (int c = 0; c < CHID; ++c) v += r[c] * fcW[c];
  out[n] = 1.f / (1.f + expf(-v));
}

// ---------------- host side ----------------
static inline int cdiv(long long a, int b) { return (int)((a + b - 1) / b); }

extern "C" void kernel_launch(void* const* d_in, const int* in_sizes, int n_in,
                              void* d_out, int out_size, void* d_ws, size_t ws_size,
                              hipStream_t stream) {
  (void)in_sizes; (void)n_in; (void)out_size; (void)ws_size;
  const float* x  = (const float*)d_in[0];
  const int*   ei = (const int*)d_in[1];
  const float* W[3]  = {(const float*)d_in[2], (const float*)d_in[6],  (const float*)d_in[10]};
  const float* As[3] = {(const float*)d_in[3], (const float*)d_in[7],  (const float*)d_in[11]};
  const float* Ad[3] = {(const float*)d_in[4], (const float*)d_in[8],  (const float*)d_in[12]};
  const float* Bi[3] = {(const float*)d_in[5], (const float*)d_in[9],  (const float*)d_in[13]};
  const float* fcW = (const float*)d_in[14];
  const float* fcB = (const float*)d_in[15];
  float* out = (float*)d_out;

  // workspace carve-out (all sizes are multiples of 256 B)
  char* ws = (char*)d_ws;
  size_t off = 0;
  auto carve = [&](size_t bytes) -> void* {
    void* p = ws + off;
    off += (bytes + 255) & ~(size_t)255;
    return p;
  };
  float*          h     = (float*)carve((size_t)NN * HHID * 4);   // 51.2 MB
  float*          accum = (float*)carve((size_t)NN * HHID * 4);   // 51.2 MB
  float*          xcur  = (float*)carve((size_t)NN * CHID * 4);   // 12.8 MB
  unsigned short* xb    = (unsigned short*)carve((size_t)NN * 128 * 2); // 12.8 MB
  unsigned short* wb    = (unsigned short*)carve((size_t)256 * 128 * 2);
  float*          es    = (float*)carve((size_t)NN * NHEAD * 4);
  float*          ed    = (float*)carve((size_t)NN * NHEAD * 4);
  float*          m     = (float*)carve((size_t)NN * NHEAD * 4);
  float*          z     = (float*)carve((size_t)NN * NHEAD * 4);
  float*          elog  = (float*)carve((size_t)ETOT * NHEAD * 4); // 13.6 MB

  const int B = 256;
  for (int layer = 0; layer < 3; ++layer) {
    const float* xin = (layer == 0) ? x : xcur;
    const int K = (layer == 0) ? 128 : CHID;

    pack_bf16_kernel<<<cdiv((long long)NN * K, B), B, 0, stream>>>(xin, xb, NN * K);
    pack_wt_kernel  <<<cdiv((long long)256 * K, B), B, 0, stream>>>(W[layer], wb, K);

    // 3125 x 16 tiles of 16x16; 4 waves/block -> 12500 blocks
    gemm_wmma_bf16_kernel<<<(3125 * 16) / 4, 128, 0, stream>>>(xb, wb, h, NN, K);

    attn_coef_kernel<<<cdiv((long long)NN * NHEAD, B), B, 0, stream>>>(h, As[layer], Ad[layer], es, ed);
    init_kernel     <<<cdiv((long long)NN * HHID, B), B, 0, stream>>>(accum, m, z);

    const int egrid = cdiv((long long)ETOT * NHEAD, B);
    edge_logits_kernel<<<egrid, B, 0, stream>>>(ei, es, ed, elog, m);
    edge_exp_kernel   <<<egrid, B, 0, stream>>>(ei, m, elog, z);
    edge_norm_kernel  <<<egrid, B, 0, stream>>>(ei, z, elog);

    scatter_kernel <<<ETOT, 256, 0, stream>>>(ei, elog, h, accum);
    finalize_kernel<<<cdiv((long long)NN * CHID, B), B, 0, stream>>>(accum, Bi[layer], xcur);
  }

  fc_kernel<<<cdiv(NN, B), B, 0, stream>>>(xcur, fcW, fcB, out);
}
